// Attention2DTAN_2061584302451
// MI455X (gfx1250) — compile-verified, matrix operand order checked by reading, not run on hardware
//
#include <hip/hip_runtime.h>

typedef __attribute__((ext_vector_type(16))) __bf16 v16bf;
typedef __attribute__((ext_vector_type(8)))  float  v8f;

// ---------------------------------------------------------------------------
// Fragment swizzle helper (CDNA5 16-bit A-matrix 16x32 layout, wave32):
//  lanes 0-15 hold M=0..15, half j -> K = (j<8 ? j : j+8)
//  lanes16-31 hold M=0..15, half j -> K = (j<8 ? j+8 : j+16)
// Inverse map used by the packers below.
// ---------------------------------------------------------------------------
__device__ __forceinline__ void frag_pos(int kk, int row16, int& lane, int& j) {
    if (kk < 8)       { lane = row16;      j = kk;      }
    else if (kk < 16) { lane = row16 + 16; j = kk - 8;  }
    else if (kk < 24) { lane = row16;      j = kk - 8;  }
    else              { lane = row16 + 16; j = kk - 16; }
}

// Pack Wk / Wv (512x512 f32, row = out channel a, col = in channel c) into
// BF16 fragments: frag[((a/16)*16 + c/32)*32 + lane]*16 + j
__global__ void prep_wkv(const float* __restrict__ Wk, const float* __restrict__ Wv,
                         __bf16* __restrict__ wkf, __bf16* __restrict__ wvf) {
    int idx = blockIdx.x * blockDim.x + threadIdx.x;   // 0..262143 = a*512 + c
    int a = idx >> 9, c = idx & 511;
    int at = a >> 4, M = a & 15;
    int chunk = c >> 5, kk = c & 31;
    int lane, j;
    frag_pos(kk, M, lane, j);
    int fi = ((at * 16 + chunk) * 32 + lane) * 16 + j;
    wkf[fi] = (__bf16)Wk[idx];
    wvf[fi] = (__bf16)Wv[idx];
}

// q[b, a, l] = textfea[b, l, :] . Wq[a, :] + bq[a]   (tiny: 0.27 GMAC)
__global__ void prep_q(const float* __restrict__ textfea, const float* __restrict__ Wq,
                       const float* __restrict__ bq, float* __restrict__ qbuf) {
    int idx = blockIdx.x * blockDim.x + threadIdx.x;   // (b*512 + a)*64 + l
    int l = idx & 63;
    int a = (idx >> 6) & 511;
    int b = idx >> 15;
    const float* t = textfea + (b * 64 + l) * 512;
    const float* w = Wq + a * 512;
    float acc = bq[a];
#pragma unroll 8
    for (int c = 0; c < 512; ++c) acc = fmaf(t[c], w[c], acc);
    qbuf[idx] = acc;
}

// ---------------------------------------------------------------------------
// Main fused kernel: one workgroup per (b, h, half-of-w).
// 8 waves; wave g computes group g's 64x32 K and V tiles with BF16 WMMA,
// then A = sum_d q*k, sigmoid, out = v*att, and the group-mean of A.
// ---------------------------------------------------------------------------
__global__ __launch_bounds__(256, 1)
void attn_main(const float* __restrict__ x,        // videofea [16,512,64,64]
               const __bf16* __restrict__ wkf,
               const __bf16* __restrict__ wvf,
               const float* __restrict__ qbuf,     // [16,512,64]
               const float* __restrict__ bk, const float* __restrict__ bv,
               float* __restrict__ out0,           // A-mean [16,1,64,64]
               float* __restrict__ out1)           // out    [16,512,64,64]
{
    __shared__ __align__(32) __bf16 ldsX[16 * 2 * 32 * 16];  // 32 KB, B-fragments
    __shared__ float meanAcc[32];

    const int h     = blockIdx.x;   // 0..63
    const int whalf = blockIdx.y;   // 0..1
    const int b     = blockIdx.z;   // 0..15
    const int tid   = threadIdx.x;
    const int lane  = tid & 31;
    const int g     = tid >> 5;     // wave id == group id
    const int rbase = (lane & 16) ? 8 : 0;

    if (tid < 32) meanAcc[tid] = 0.0f;

    // Stage X = videofea[b, :, h, whalf*32 .. +31] as BF16 B-fragments in LDS.
    const float* xbase = x + ((b * 512) * 64 + h) * 64 + whalf * 32;
    for (int i = tid; i < 512 * 32; i += 256) {
        int c = i >> 5, wc = i & 31;
        float v = xbase[c * 4096 + wc];
        int chunk = c >> 5, kk = c & 31;
        int ntile = wc >> 4, N = wc & 15;
        int ln, lj;
        frag_pos(kk, N, ln, lj);
        ldsX[((chunk * 2 + ntile) * 32 + ln) * 16 + lj] = (__bf16)v;
    }
    __syncthreads();

    // Accumulators, biases folded into init (C/D layout: reg r -> row rbase+r).
    v8f accK[4][2], accV[4][2];
#pragma unroll
    for (int mi = 0; mi < 4; ++mi)
#pragma unroll
        for (int r = 0; r < 8; ++r) {
            int d = mi * 16 + rbase + r;
            float kb = bk[g * 64 + d];
            float vb = bv[g * 64 + d];
#pragma unroll
            for (int ni = 0; ni < 2; ++ni) { accK[mi][ni][r] = kb; accV[mi][ni][r] = vb; }
        }

    // K = Wk_g @ X, V = Wv_g @ X : 16 k-steps of K=32, 16 WMMAs each.
    for (int cs = 0; cs < 16; ++cs) {
        v16bf xb0 = *(const v16bf*)&ldsX[((cs * 2 + 0) * 32 + lane) * 16];
        v16bf xb1 = *(const v16bf*)&ldsX[((cs * 2 + 1) * 32 + lane) * 16];
#pragma unroll
        for (int mi = 0; mi < 4; ++mi) {
            int at = g * 4 + mi;
            v16bf wa = *(const v16bf*)(wkf + (((at * 16 + cs) * 32 + lane) * 16));
            v16bf va = *(const v16bf*)(wvf + (((at * 16 + cs) * 32 + lane) * 16));
            accK[mi][0] = __builtin_amdgcn_wmma_f32_16x16x32_bf16(false, wa, false, xb0,
                              (short)0, accK[mi][0], false, false);
            accK[mi][1] = __builtin_amdgcn_wmma_f32_16x16x32_bf16(false, wa, false, xb1,
                              (short)0, accK[mi][1], false, false);
            accV[mi][0] = __builtin_amdgcn_wmma_f32_16x16x32_bf16(false, va, false, xb0,
                              (short)0, accV[mi][0], false, false);
            accV[mi][1] = __builtin_amdgcn_wmma_f32_16x16x32_bf16(false, va, false, xb1,
                              (short)0, accV[mi][1], false, false);
        }
    }

    // A[g,h,w] = sum_d q[b,g,d,h] * K[d,w] : in-register reduce + xor-16 shuffle.
    const float* qb = qbuf + (b * 512 + g * 64) * 64 + h;   // + d*64
    float p0 = 0.0f, p1 = 0.0f;
#pragma unroll
    for (int mi = 0; mi < 4; ++mi)
#pragma unroll
        for (int r = 0; r < 8; ++r) {
            int d = mi * 16 + rbase + r;
            float qv = qb[d * 64];
            p0 = fmaf(qv, accK[mi][0][r], p0);
            p1 = fmaf(qv, accK[mi][1][r], p1);
        }
    p0 += __shfl_xor(p0, 16, 32);
    p1 += __shfl_xor(p1, 16, 32);
    float att0 = 1.0f / (1.0f + __expf(-p0));
    float att1 = 1.0f / (1.0f + __expf(-p1));

    // Group-mean of (pre-sigmoid) A via LDS float atomics (ds_add_f32).
    if (lane < 16) {
        atomicAdd(&meanAcc[lane],      p0);
        atomicAdd(&meanAcc[16 + lane], p1);
    }

    // out[b, g*64+d, h, w] = V[d,w] * sigmoid(A[g,h,w])
    const int wbase = whalf * 32;
#pragma unroll
    for (int mi = 0; mi < 4; ++mi)
#pragma unroll
        for (int ni = 0; ni < 2; ++ni) {
            float a_t = (ni == 0) ? att0 : att1;
            int wcol = wbase + ni * 16 + (lane & 15);
            float* o = out1 + ((b * 512 + g * 64 + mi * 16 + rbase) * 64 + h) * 64 + wcol;
#pragma unroll
            for (int r = 0; r < 8; ++r)
                o[r * 4096] = accV[mi][ni][r] * a_t;
        }

    __syncthreads();
    if (tid < 32)
        out0[(b * 64 + h) * 64 + wbase + tid] = meanAcc[tid] * 0.125f;
}

extern "C" void kernel_launch(void* const* d_in, const int* in_sizes, int n_in,
                              void* d_out, int out_size, void* d_ws, size_t ws_size,
                              hipStream_t stream) {
    const float* videofea = (const float*)d_in[0];
    const float* textfea  = (const float*)d_in[1];
    const float* Wq       = (const float*)d_in[2];
    const float* bq       = (const float*)d_in[3];
    const float* Wk       = (const float*)d_in[4];
    const float* bk       = (const float*)d_in[5];
    const float* Wv       = (const float*)d_in[6];
    const float* bv       = (const float*)d_in[7];
    float* out = (float*)d_out;

    char* ws = (char*)d_ws;
    __bf16* wkf  = (__bf16*)ws;                       // 512 KB
    __bf16* wvf  = (__bf16*)(ws + (512 << 10));       // 512 KB
    float*  qbuf = (float*)(ws + (1 << 20));          // 2 MB

    prep_wkv<<<262144 / 256, 256, 0, stream>>>(Wk, Wv, wkf, wvf);
    prep_q  <<<524288 / 256, 256, 0, stream>>>(textfea, Wq, bq, qbuf);

    dim3 grid(64, 2, 16);                             // (h, whalf, b)
    attn_main<<<grid, 256, 0, stream>>>(videofea, wkf, wvf, qbuf, bk, bv,
                                        out, out + 65536);
}
